// GraphGCN_68865505624353
// MI455X (gfx1250) — matrix operand order repaired and checked by reading, not compiled, not run on hardware
//
#include <hip/hip_runtime.h>

#define N_NODES  50000
#define N_EDGES  800000
#define CH       128
#define OUTC     8
#define N_GRAPHS 64

typedef __attribute__((ext_vector_type(16))) __bf16 v16bf;
typedef __attribute__((ext_vector_type(8)))  __bf16 v8bf;
typedef __attribute__((ext_vector_type(8)))  float  v8f;

// ---------- degree / normalization ----------
__global__ __launch_bounds__(256) void k_fill1(float* p, int n) {
    int i = blockIdx.x * 256 + threadIdx.x;
    if (i < n) p[i] = 1.0f;                       // self-loop contributes 1
}

__global__ __launch_bounds__(256) void k_deg(const int* __restrict__ dst, float* __restrict__ deg) {
    int e = blockIdx.x * 256 + threadIdx.x;
    if (e < N_EDGES) atomicAdd(&deg[dst[e]], 1.0f);
}

__global__ __launch_bounds__(256) void k_rsqrt_inplace(float* p, int n) {
    int i = blockIdx.x * 256 + threadIdx.x;
    if (i < n) p[i] = rsqrtf(p[i]);               // deg >= 1 always
}

// ---------- f32 -> bf16 conversion (activations) ----------
__global__ __launch_bounds__(256) void k_cvt_bf16(const float* __restrict__ in,
                                                  __bf16* __restrict__ out, int n) {
    int i = blockIdx.x * 256 + threadIdx.x;
    if (i < n) out[i] = (__bf16)in[i];
}

// ---------- pack W [128,128] f32 into per-lane WMMA B-fragment layout (bf16) ----------
// Packed layout, region per (ntile t, kchunk kk) = 512 bf16 (1024 B):
//   first 256 elems  : lanes 0..31 x fragment elements 0..7  (lane*8 contiguous)
//   second 256 elems : lanes 0..31 x fragment elements 8..15
// Fragment semantics: b[i] = W[(kk*32 + (i>>3)*16 + g*8 + (i&7))*CH + t*16 + (lane&15)], g = lane>>4.
__global__ __launch_bounds__(256) void k_pack_w(const float* __restrict__ W,
                                                __bf16* __restrict__ P) {
    int idx  = blockIdx.x * 256 + threadIdx.x;    // 0..16383
    int i    = idx & 15;
    int lane = (idx >> 4) & 31;
    int kk   = (idx >> 9) & 3;
    int t    = idx >> 11;
    int g    = lane >> 4;
    int col  = t * 16 + (lane & 15);
    int k    = kk * 32 + (i >> 3) * 16 + g * 8 + (i & 7);
    int pofs = (t * 4 + kk) * 512 + (i >> 3) * 256 + lane * 8 + (i & 7);
    P[pofs] = (__bf16)W[k * CH + col];
}

// ---------- WMMA GEMM: C[M,128] = A[M,128] @ W[128,128], bf16 in / f32 out ----------
// 256 threads = 8 waves; block owns a 16-row stripe, wave w owns n-tile w.
// K=128 -> 4x v_wmma_f32_16x16x32_bf16; all operand loads are coalesced b128.
__global__ __launch_bounds__(256) void k_gemm_bf16(const __bf16* __restrict__ A,
                                                   const __bf16* __restrict__ Bp,
                                                   float* __restrict__ C) {
    const int lane = threadIdx.x & 31;
    const int wave = threadIdx.x >> 5;            // n-tile 0..7
    const int g    = lane >> 4;                   // lane half
    const int l    = lane & 15;
    const int row  = blockIdx.x * 16 + l;         // A row held by this lane

    const __bf16* arow  = A + (size_t)row * CH;
    const __bf16* bfrag = Bp + (size_t)wave * 4 * 512;

    v8f acc = {};
    #pragma unroll
    for (int kk = 0; kk < 4; ++kk) {
        const int kb = kk * 32;
        // A 16x32 fragment: lane half g holds K {kb+g*8..+7, kb+16+g*8..+7} (two b128s)
        v8bf alo = *(const v8bf*)(arow + kb + g * 8);
        v8bf ahi = *(const v8bf*)(arow + kb + 16 + g * 8);
        // B fragment: pre-packed, two coalesced b128s
        v8bf blo = *(const v8bf*)(bfrag + kk * 512 + lane * 8);
        v8bf bhi = *(const v8bf*)(bfrag + kk * 512 + 256 + lane * 8);
        v16bf a = __builtin_shufflevector(alo, ahi, 0,1,2,3,4,5,6,7,8,9,10,11,12,13,14,15);
        v16bf b = __builtin_shufflevector(blo, bhi, 0,1,2,3,4,5,6,7,8,9,10,11,12,13,14,15);
        acc = __builtin_amdgcn_wmma_f32_16x16x32_bf16(false, a, false, b,
                                                      (short)0, acc, false, false);
    }
    // C/D layout: VGPR r -> row g*8+r, column = lane&15 within the wave's n-tile.
    const int col = wave * 16 + l;
    #pragma unroll
    for (int r = 0; r < 8; ++r) {
        C[(size_t)(blockIdx.x * 16 + g * 8 + r) * CH + col] = acc[r];
    }
}

// ---------- self-loop init: agg = H * dinv[v]^2 ----------
__global__ __launch_bounds__(256) void k_self_init(const float* __restrict__ H,
                                                   const float* __restrict__ dinv,
                                                   float* __restrict__ agg) {
    int i = blockIdx.x * 256 + threadIdx.x;       // over N*CH (exact multiple)
    int v = i >> 7;
    float d = dinv[v];
    agg[i] = H[i] * d * d;
}

// ---------- per-edge message passing: one wave per edge ----------
__global__ __launch_bounds__(256) void k_edge_mp(const int* __restrict__ src,
                                                 const int* __restrict__ dst,
                                                 const float* __restrict__ dinv,
                                                 const float* __restrict__ H,
                                                 float* __restrict__ agg) {
    const int lane = threadIdx.x & 31;
    const int e = blockIdx.x * 8 + (threadIdx.x >> 5);
    const int s = src[e], d = dst[e];
    const float norm = dinv[s] * dinv[d];
    const float4 hv = ((const float4*)(H + (size_t)s * CH))[lane];  // 128 ch / wave
    float* out = agg + (size_t)d * CH + lane * 4;
    atomicAdd(out + 0, hv.x * norm);
    atomicAdd(out + 1, hv.y * norm);
    atomicAdd(out + 2, hv.z * norm);
    atomicAdd(out + 3, hv.w * norm);
}

// ---------- epilogue layer 1: bias + ReLU + bf16 for next GEMM ----------
__global__ __launch_bounds__(256) void k_relu_bias_bf16(const float* __restrict__ agg,
                                                        const float* __restrict__ bias,
                                                        __bf16* __restrict__ out) {
    int i = blockIdx.x * 256 + threadIdx.x;
    float v = agg[i] + bias[i & (CH - 1)];
    out[i] = (__bf16)fmaxf(v, 0.0f);
}

__global__ __launch_bounds__(256) void k_zero(float* p, int n) {
    int i = blockIdx.x * 256 + threadIdx.x;
    if (i < n) p[i] = 0.0f;
}

// ---------- epilogue layer 2: bias + ReLU fused with mean-pool accumulation ----------
__global__ __launch_bounds__(256) void k_relu_pool(const float* __restrict__ agg,
                                                   const float* __restrict__ bias,
                                                   const int* __restrict__ batch,
                                                   float* __restrict__ pool,
                                                   float* __restrict__ cnt) {
    int i = blockIdx.x * 256 + threadIdx.x;
    int v = i >> 7, c = i & (CH - 1);
    float val = fmaxf(agg[i] + bias[c], 0.0f);
    int b = batch[v];
    atomicAdd(&pool[b * CH + c], val);
    if (c == 0) atomicAdd(&cnt[b], 1.0f);
}

// ---------- final head: [64,128] -> [64,8] ----------
__global__ __launch_bounds__(512) void k_final(const float* __restrict__ pool,
                                               const float* __restrict__ cnt,
                                               const float* __restrict__ Wlin,
                                               const float* __restrict__ blin,
                                               float* __restrict__ out) {
    int t = threadIdx.x;                          // 0..511 = 64 graphs x 8 outputs
    int gph = t >> 3, o = t & 7;
    float inv = 1.0f / fmaxf(cnt[gph], 1.0f);
    float s = blin[o];
    #pragma unroll 4
    for (int c = 0; c < CH; ++c)
        s += pool[gph * CH + c] * inv * Wlin[c * OUTC + o];
    out[t] = s;
}

extern "C" void kernel_launch(void* const* d_in, const int* in_sizes, int n_in,
                              void* d_out, int out_size, void* d_ws, size_t ws_size,
                              hipStream_t stream) {
    (void)in_sizes; (void)n_in; (void)out_size; (void)ws_size;
    const float* x     = (const float*)d_in[0];
    const int*   ei    = (const int*)  d_in[1];
    const int*   batch = (const int*)  d_in[2];
    /* d_in[3] = num_graphs scalar (compile-time constant here) */
    const float* W1    = (const float*)d_in[4];
    const float* b1    = (const float*)d_in[5];
    const float* W2    = (const float*)d_in[6];
    const float* b2    = (const float*)d_in[7];
    const float* Wlin  = (const float*)d_in[8];
    const float* blin  = (const float*)d_in[9];
    float* out = (float*)d_out;

    const int* src = ei;
    const int* dst = ei + N_EDGES;

    // workspace carve-out (256B aligned regions), ~64 MB total
    char* ws = (char*)d_ws;
    size_t off = 0;
    auto carve = [&](size_t bytes) {
        char* p = ws + off;
        off += (bytes + 255) & ~(size_t)255;
        return p;
    };
    __bf16* bufb = (__bf16*)carve((size_t)N_NODES * CH * 2);  // X bf16, then act1 bf16
    __bf16* W1p  = (__bf16*)carve((size_t)CH * CH * 2);       // packed WMMA fragments
    __bf16* W2p  = (__bf16*)carve((size_t)CH * CH * 2);
    float*  H    = (float*) carve((size_t)N_NODES * CH * 4);  // GEMM output (both layers)
    float*  agg  = (float*) carve((size_t)N_NODES * CH * 4);  // scatter accumulator
    float*  dinv = (float*) carve((size_t)N_NODES * 4);
    float*  pool = (float*) carve((size_t)N_GRAPHS * CH * 4); // cnt follows contiguously
    float*  cnt  = (float*) carve((size_t)N_GRAPHS * 4);
    (void)cnt; cnt = pool + N_GRAPHS * CH;                    // pool region is 256B-multiple

    const int NC     = N_NODES * CH;       // 6,400,000 (exact multiple of 256)
    const int gridNC = NC / 256;           // 25000
    const int gridN  = (N_NODES + 255) / 256;
    const int gridE  = (N_EDGES + 255) / 256;

    // symmetric normalization
    k_fill1        <<<gridN, 256, 0, stream>>>(dinv, N_NODES);
    k_deg          <<<gridE, 256, 0, stream>>>(dst, dinv);
    k_rsqrt_inplace<<<gridN, 256, 0, stream>>>(dinv, N_NODES);

    // bf16 staging: activations + fragment-packed weights
    k_cvt_bf16<<<gridNC, 256, 0, stream>>>(x, bufb, NC);
    k_pack_w  <<<CH * CH / 256, 256, 0, stream>>>(W1, W1p);
    k_pack_w  <<<CH * CH / 256, 256, 0, stream>>>(W2, W2p);

    // ----- layer 1 -----
    k_gemm_bf16     <<<N_NODES / 16, 256, 0, stream>>>(bufb, W1p, H);
    k_self_init     <<<gridNC,       256, 0, stream>>>(H, dinv, agg);
    k_edge_mp       <<<N_EDGES / 8,  256, 0, stream>>>(src, dst, dinv, H, agg);
    k_relu_bias_bf16<<<gridNC,       256, 0, stream>>>(agg, b1, bufb);

    // ----- layer 2 -----
    k_gemm_bf16<<<N_NODES / 16, 256, 0, stream>>>(bufb, W2p, H);
    k_self_init<<<gridNC,       256, 0, stream>>>(H, dinv, agg);
    k_edge_mp  <<<N_EDGES / 8,  256, 0, stream>>>(src, dst, dinv, H, agg);

    // ----- pooling + head -----
    k_zero     <<<(N_GRAPHS * CH + N_GRAPHS + 255) / 256, 256, 0, stream>>>(pool, N_GRAPHS * CH + N_GRAPHS);
    k_relu_pool<<<gridNC, 256, 0, stream>>>(agg, b2, batch, pool, cnt);
    k_final    <<<1, 512, 0, stream>>>(pool, cnt, Wlin, blin, out);
}
